// DHSMLanguageModel_37014028157375
// MI455X (gfx1250) — matrix operand order repaired and checked by reading, not compiled, not run on hardware
//
#include <hip/hip_runtime.h>
#include <hip/hip_bf16.h>

// ---------------- model dims (fixed by the reference) ----------------
#define BB   32
#define SS   256
#define VV   10000
#define DD   768
#define NN   128
#define LL   6
#define MM   (BB * SS)          // 8192 token rows
#define CLAMP_V 10.0f
#define EPS_V   1e-5f

typedef __hip_bfloat16 hbf;                                  // storage type
typedef __bf16 v16bf __attribute__((ext_vector_type(16)));   // WMMA operand
typedef float  v8f   __attribute__((ext_vector_type(8)));    // WMMA accumulator

// ---------------------------------------------------------------------
// WMMA fragment loaders (CDNA5 16x16x32 bf16 layouts, ISA 7.12.2):
//  A 16x32 (MxK): lane L -> row m0+(L&15); elems 0..7  = K[k0+8*(L>>4) .. +7],
//                 elems 8..15 = K[k0+16+8*(L>>4) .. +7]   (two 16B chunks)
//  B 32x16 (KxN): lane L -> col n0+(L&15); elems 0..7  = K[k0+8*(L>>4) .. +7],
//                 elems 8..15 = K[k0+16+8*(L>>4) .. +7]   (two 16B chunks of W[n])
// ---------------------------------------------------------------------
__device__ __forceinline__ v16bf load_frag_a(const hbf* __restrict__ X, int ldx,
                                             int m0, int k0, int lane) {
  const int m    = m0 + (lane & 15);
  const int half = lane >> 4;
  const hbf* p = X + (size_t)m * ldx + k0 + half * 8;
  union { uint4 u[2]; v16bf v; } f;
  f.u[0] = *reinterpret_cast<const uint4*>(p);
  f.u[1] = *reinterpret_cast<const uint4*>(p + 16);
  return f.v;
}

__device__ __forceinline__ v16bf load_frag_b(const hbf* __restrict__ W, int ldw,
                                             int n0, int k0, int lane, int nmax) {
  int n = n0 + (lane & 15);
  if (n >= nmax) n = nmax - 1;                        // edge clamp (head GEMM)
  const int kh = lane >> 4;
  const hbf* p = W + (size_t)n * ldw + k0 + kh * 8;
  union { uint4 u[2]; v16bf v; } f;
  f.u[0] = *reinterpret_cast<const uint4*>(p);
  f.u[1] = *reinterpret_cast<const uint4*>(p + 16);
  return f.v;
}

// ---------------------------------------------------------------------
// Software-pipelined bf16 GEMM:  Y[M,Nout] (+)= X[M,K] * W[Nout,K]^T (+bias)
// Block = 128 threads (4 waves). Wave tile = 32M x 64N (2x4 WMMA tiles,
// 8 WMMA per 32-wide K step). Two K-stages double-buffered in registers so
// global fragment loads overlap the WMMAs of the previous stage (everything
// is L2-resident: ~100MB working set vs 192MB L2).
// Requires: M % 128 == 0, K % 64 == 0 (true for all call sites).
// ---------------------------------------------------------------------
__global__ __launch_bounds__(128) void
gemm_bf16_wmma(const hbf* __restrict__ X, const hbf* __restrict__ W,
               float* __restrict__ Y, int M, int Nout, int K,
               const float* __restrict__ bias, int accumulate) {
  const int lane = threadIdx.x & 31;
  const int wave = threadIdx.x >> 5;
  const int m0 = blockIdx.x * 128 + wave * 32;
  const int n0 = blockIdx.y * 64;

  v8f acc[2][4] = {};
  v16bf a0[2], b0[4], a1[2], b1[4];

  // prologue: stage 0 = k 0..31
  a0[0] = load_frag_a(X, K, m0,      0, lane);
  a0[1] = load_frag_a(X, K, m0 + 16, 0, lane);
#pragma unroll
  for (int t = 0; t < 4; ++t) b0[t] = load_frag_b(W, K, n0 + 16 * t, 0, lane, Nout);

  for (int k = 0; k < K; k += 64) {
    // issue stage-1 loads (k+32) before consuming stage 0
    a1[0] = load_frag_a(X, K, m0,      k + 32, lane);
    a1[1] = load_frag_a(X, K, m0 + 16, k + 32, lane);
#pragma unroll
    for (int t = 0; t < 4; ++t) b1[t] = load_frag_b(W, K, n0 + 16 * t, k + 32, lane, Nout);

#pragma unroll
    for (int mt = 0; mt < 2; ++mt)
#pragma unroll
      for (int t = 0; t < 4; ++t)
        acc[mt][t] = __builtin_amdgcn_wmma_f32_16x16x32_bf16(
            false, a0[mt], false, b0[t], (short)0, acc[mt][t], false, false);

    if (k + 64 < K) {  // issue stage-0 loads for k+64 before consuming stage 1
      a0[0] = load_frag_a(X, K, m0,      k + 64, lane);
      a0[1] = load_frag_a(X, K, m0 + 16, k + 64, lane);
#pragma unroll
      for (int t = 0; t < 4; ++t) b0[t] = load_frag_b(W, K, n0 + 16 * t, k + 64, lane, Nout);
    }

#pragma unroll
    for (int mt = 0; mt < 2; ++mt)
#pragma unroll
      for (int t = 0; t < 4; ++t)
        acc[mt][t] = __builtin_amdgcn_wmma_f32_16x16x32_bf16(
            false, a1[mt], false, b1[t], (short)0, acc[mt][t], false, false);
  }

  // D layout: VGPR r, lanes 0-15 -> M=mt0+r, N=lane; lanes 16-31 -> M=mt0+8+r
  const int ncol = lane & 15;
#pragma unroll
  for (int mt = 0; mt < 2; ++mt) {
    const int mrow = m0 + mt * 16 + (lane >> 4) * 8;
#pragma unroll
    for (int t = 0; t < 4; ++t) {
      const int n = n0 + t * 16 + ncol;
      if (n < Nout) {
        const float bv = bias ? bias[n] : 0.0f;
#pragma unroll
        for (int r = 0; r < 8; ++r) {
          const size_t idx = (size_t)(mrow + r) * Nout + n;
          float v = acc[mt][t][r] + bv;
          if (accumulate) v += Y[idx];
          Y[idx] = v;
        }
      }
    }
  }
}

// ---------------------------------------------------------------------
// Sequential SSM scan: state = clip(state @ A^T + Bx[s], +-10), per batch.
// One block (one WGP) per batch, 128 threads; thread n owns output element n
// and keeps its entire A row (128 floats = 32 float4) in VGPRs, so each step
// is 32 broadcast ds_load_b128 of the shared state + 128 reg-resident FMAs.
// Next step's Bx row is prefetched during the FMA chain. One barrier/step
// via double-buffered state.
// ---------------------------------------------------------------------
__global__ __launch_bounds__(128) void
ssm_scan(const float* __restrict__ A, const float* __restrict__ Bx,
         hbf* __restrict__ StH) {
  __shared__ __align__(16) float st[2][NN];
  const int b = blockIdx.x;
  const int n = threadIdx.x;        // 0..127

  float4 arow[NN / 4];              // A[n][0..127] in registers
  const float4* Ar = reinterpret_cast<const float4*>(A + (size_t)n * NN);
#pragma unroll
  for (int q = 0; q < NN / 4; ++q) arow[q] = Ar[q];

  st[0][n] = 0.0f;
  float bxv = Bx[((size_t)b * SS) * NN + n];          // s = 0
  __syncthreads();

  int cur = 0;
  for (int s = 0; s < SS; ++s) {
    float bxn = 0.0f;
    if (s + 1 < SS) bxn = Bx[((size_t)b * SS + s + 1) * NN + n];  // prefetch
    const float4* st4 = reinterpret_cast<const float4*>(st[cur]);
    float a = bxv;
#pragma unroll
    for (int q = 0; q < NN / 4; ++q) {
      const float4 sv = st4[q];
      const float4 av = arow[q];
      a = fmaf(sv.x, av.x, a);
      a = fmaf(sv.y, av.y, a);
      a = fmaf(sv.z, av.z, a);
      a = fmaf(sv.w, av.w, a);
    }
    a = fminf(fmaxf(a, -CLAMP_V), CLAMP_V);
    st[cur ^ 1][n] = a;
    StH[((size_t)b * SS + s) * NN + n] = __float2bfloat16(a);
    bxv = bxn;
    cur ^= 1;
    __syncthreads();
  }
}

// ---------------------------------------------------------------------
// Embedding + positional add; writes fp32 x and bf16 xh.
// ---------------------------------------------------------------------
__global__ void embed_kernel(const int* __restrict__ ids, const float* __restrict__ emb,
                             const float* __restrict__ pos, float* __restrict__ x,
                             hbf* __restrict__ xh) {
  const int m = blockIdx.x;               // m = b*SS + s
  const int s = m % SS;
  const int id = ids[m];
  for (int d = threadIdx.x; d < DD; d += blockDim.x) {
    const float v = emb[(size_t)id * DD + d] + pos[(size_t)s * DD + d];
    x[(size_t)m * DD + d]  = v;
    xh[(size_t)m * DD + d] = __float2bfloat16(v);
  }
}

// ---------------------------------------------------------------------
// Fused epilogue per row: gate logits (2 dots over D), softmax(2),
// out = g0*raw + g1*x, LayerNorm -> overwrite x (fp32) and xh (bf16).
// 256 threads/row, D = 768 = 3*256.
// ---------------------------------------------------------------------
__global__ void fuse_gate_ln(const float* __restrict__ raw, float* __restrict__ x,
                             hbf* __restrict__ xh, const float* __restrict__ gw,
                             const float* __restrict__ gb, const float* __restrict__ lnw,
                             const float* __restrict__ lnb) {
  __shared__ float red[256];
  __shared__ float sc[2];
  const int m = blockIdx.x;
  const int t = threadIdx.x;
  const float* xr = x   + (size_t)m * DD;
  const float* rr = raw + (size_t)m * DD;

  float xv[3], d0 = 0.f, d1 = 0.f;
#pragma unroll
  for (int i = 0; i < 3; ++i) {
    const int d = t + i * 256;
    const float v = xr[d];
    xv[i] = v;
    d0 += v * gw[d];
    d1 += v * gw[DD + d];
  }
  red[t] = d0; __syncthreads();
  for (int o = 128; o > 0; o >>= 1) { if (t < o) red[t] += red[t + o]; __syncthreads(); }
  if (t == 0) sc[0] = red[0] + gb[0];
  __syncthreads();
  red[t] = d1; __syncthreads();
  for (int o = 128; o > 0; o >>= 1) { if (t < o) red[t] += red[t + o]; __syncthreads(); }
  if (t == 0) sc[1] = red[0] + gb[1];
  __syncthreads();

  const float mx = fmaxf(sc[0], sc[1]);
  const float e0 = __expf(sc[0] - mx), e1 = __expf(sc[1] - mx);
  const float inv = 1.0f / (e0 + e1);
  const float g0 = e0 * inv, g1 = e1 * inv;

  float vals[3], lsum = 0.f;
#pragma unroll
  for (int i = 0; i < 3; ++i) {
    const int d = t + i * 256;
    const float v = g0 * rr[d] + g1 * xv[i];
    vals[i] = v; lsum += v;
  }
  __syncthreads();
  red[t] = lsum; __syncthreads();
  for (int o = 128; o > 0; o >>= 1) { if (t < o) red[t] += red[t + o]; __syncthreads(); }
  const float mu = red[0] * (1.0f / DD);
  __syncthreads();
  float vs = 0.f;
#pragma unroll
  for (int i = 0; i < 3; ++i) { const float dv = vals[i] - mu; vs += dv * dv; }
  red[t] = vs; __syncthreads();
  for (int o = 128; o > 0; o >>= 1) { if (t < o) red[t] += red[t + o]; __syncthreads(); }
  const float rstd = rsqrtf(red[0] * (1.0f / DD) + EPS_V);
#pragma unroll
  for (int i = 0; i < 3; ++i) {
    const int d = t + i * 256;
    const float y = (vals[i] - mu) * rstd * lnw[d] + lnb[d];
    x[(size_t)m * DD + d]  = y;
    xh[(size_t)m * DD + d] = __float2bfloat16(y);
  }
}

// Final LayerNorm -> bf16 only (feeds the head GEMM).
__global__ void final_ln(const float* __restrict__ x, const float* __restrict__ w,
                         const float* __restrict__ b, hbf* __restrict__ xh) {
  __shared__ float red[256];
  const int m = blockIdx.x;
  const int t = threadIdx.x;
  const float* xr = x + (size_t)m * DD;
  float vals[3], lsum = 0.f;
#pragma unroll
  for (int i = 0; i < 3; ++i) { vals[i] = xr[t + i * 256]; lsum += vals[i]; }
  red[t] = lsum; __syncthreads();
  for (int o = 128; o > 0; o >>= 1) { if (t < o) red[t] += red[t + o]; __syncthreads(); }
  const float mu = red[0] * (1.0f / DD);
  __syncthreads();
  float vs = 0.f;
#pragma unroll
  for (int i = 0; i < 3; ++i) { const float dv = vals[i] - mu; vs += dv * dv; }
  red[t] = vs; __syncthreads();
  for (int o = 128; o > 0; o >>= 1) { if (t < o) red[t] += red[t + o]; __syncthreads(); }
  const float rstd = rsqrtf(red[0] * (1.0f / DD) + EPS_V);
#pragma unroll
  for (int i = 0; i < 3; ++i) {
    const int d = t + i * 256;
    xh[(size_t)m * DD + d] = __float2bfloat16((vals[i] - mu) * rstd * w[d] + b[d]);
  }
}

// fp32 -> bf16 weight conversion.
__global__ void cvt_bf16(const float* __restrict__ src, hbf* __restrict__ dst, size_t n) {
  const size_t i = (size_t)blockIdx.x * blockDim.x + threadIdx.x;
  if (i < n) dst[i] = __float2bfloat16(src[i]);
}

// ---------------------------------------------------------------------
extern "C" void kernel_launch(void* const* d_in, const int* in_sizes, int n_in,
                              void* d_out, int out_size, void* d_ws, size_t ws_size,
                              hipStream_t stream) {
  const int*   ids   = (const int*)  d_in[0];
  const float* emb   = (const float*)d_in[1];
  const float* pos   = (const float*)d_in[2];
  const float* A     = (const float*)d_in[3];
  const float* Bw    = (const float*)d_in[4];
  const float* Cw    = (const float*)d_in[5];
  const float* Dw    = (const float*)d_in[6];
  const float* gw    = (const float*)d_in[7];
  const float* gb    = (const float*)d_in[8];
  const float* lnw   = (const float*)d_in[9];
  const float* lnb   = (const float*)d_in[10];
  const float* nw    = (const float*)d_in[11];
  const float* nb    = (const float*)d_in[12];
  const float* headw = (const float*)d_in[13];
  const float* headb = (const float*)d_in[14];
  float* logits = (float*)d_out;

  // workspace carve-up (~94 MB total)
  char* ws = (char*)d_ws;
  size_t off = 0;
  auto carve = [&](size_t bytes) -> void* {
    void* p = ws + off;
    off += (bytes + 255) & ~(size_t)255;
    return p;
  };
  float* x    = (float*)carve((size_t)MM * DD * 4);
  hbf*   xh   = (hbf*)  carve((size_t)MM * DD * 2);
  float* raw  = (float*)carve((size_t)MM * DD * 4);
  float* bx   = (float*)carve((size_t)MM * NN * 4);   // [B,S,N]
  hbf*   stH  = (hbf*)  carve((size_t)MM * NN * 2);   // [B*S,N] bf16
  hbf*   BwH  = (hbf*)  carve((size_t)LL * NN * DD * 2);
  hbf*   CwH  = (hbf*)  carve((size_t)LL * DD * NN * 2);
  hbf*   DwH  = (hbf*)  carve((size_t)LL * DD * DD * 2);
  hbf*   headH= (hbf*)  carve((size_t)VV * DD * 2);

  auto cvt = [&](const float* s, hbf* d, size_t n) {
    cvt_bf16<<<dim3((unsigned)((n + 255) / 256)), dim3(256), 0, stream>>>(s, d, n);
  };
  cvt(Bw,    BwH,  (size_t)LL * NN * DD);
  cvt(Cw,    CwH,  (size_t)LL * DD * NN);
  cvt(Dw,    DwH,  (size_t)LL * DD * DD);
  cvt(headw, headH, (size_t)VV * DD);

  embed_kernel<<<dim3(MM), dim3(256), 0, stream>>>(ids, emb, pos, x, xh);

  const dim3 gemm_block(128);
  for (int l = 0; l < LL; ++l) {
    // Bx[B,S,N] = x @ Bw^T   (M=8192, Nout=128, K=768)
    gemm_bf16_wmma<<<dim3(MM / 128, NN / 64), gemm_block, 0, stream>>>(
        xh, BwH + (size_t)l * NN * DD, bx, MM, NN, DD, nullptr, 0);
    // raw = x @ Dw^T         (M=8192, Nout=768, K=768) -- depends only on x
    gemm_bf16_wmma<<<dim3(MM / 128, DD / 64), gemm_block, 0, stream>>>(
        xh, DwH + (size_t)l * DD * DD, raw, MM, DD, DD, nullptr, 0);
    // sequential scan (fp32, clip +-10) -> bf16 states
    ssm_scan<<<dim3(BB), dim3(NN), 0, stream>>>(A + (size_t)l * NN * NN, bx, stH);
    // raw += states @ Cw^T   (M=8192, Nout=768, K=128)
    gemm_bf16_wmma<<<dim3(MM / 128, DD / 64), gemm_block, 0, stream>>>(
        stH, CwH + (size_t)l * DD * NN, raw, MM, DD, NN, nullptr, 1);
    // gate + softmax + mix + LayerNorm, in-place on x / xh
    fuse_gate_ln<<<dim3(MM), dim3(256), 0, stream>>>(
        raw, x, xh, gw + (size_t)l * 2 * DD, gb + (size_t)l * 2,
        lnw + (size_t)l * DD, lnb + (size_t)l * DD);
  }

  final_ln<<<dim3(MM), dim3(256), 0, stream>>>(x, nw, nb, xh);

  // logits = x @ head_w^T + head_b   (M=8192, Nout=10000, K=768)
  gemm_bf16_wmma<<<dim3(MM / 128, (VV + 63) / 64), gemm_block, 0, stream>>>(
      xh, headH, logits, MM, VV, DD, headb, 0);
}
